// NoisyTopkRouter_5506148073581
// MI455X (gfx1250) — matrix operand order repaired and checked by reading, not compiled, not run on hardware
//
#include <hip/hip_runtime.h>

typedef __attribute__((ext_vector_type(16))) __bf16 v16bf;
typedef __attribute__((ext_vector_type(4)))  __bf16 v4bf;
typedef __attribute__((ext_vector_type(8)))  float  v8f;

#define Bn   4
#define Sn   4096
#define Dn   4096
#define En   64
#define Kn   8
#define MTOT (Bn * Sn)          // 16384 tokens
#define KC   128                // K-chunk staged in LDS
#define NCHUNK (Dn / KC)        // 32
#define WAVES   4
#define THREADS (WAVES * 32)    // 128
#define ROWS    (WAVES * 16)    // 64 tokens per block
#define NBLOCKS (MTOT / ROWS)   // 256
#define WBYTES  ((size_t)2 * En * Dn * sizeof(__bf16))   // 1 MB bf16 weights

// ---------------------------------------------------------------------------
// Shared epilogue: per-token top-8 + masked softmax (one thread per token).
// ---------------------------------------------------------------------------
static __device__ __forceinline__
void topk_softmax_row(const float* rowL, int gm,
                      float* __restrict__ outP, float* __restrict__ outI)
{
    float tv[8];
    int   ti[8];
#pragma unroll
    for (int i = 0; i < 8; ++i) { tv[i] = -__builtin_inff(); ti[i] = -1; }

    for (int n = 0; n < En; ++n) {
        float cv = rowL[n];
        int   ci = n;
#pragma unroll
        for (int i = 0; i < 8; ++i) {
            if (cv > tv[i]) {
                float tf = tv[i]; int tj = ti[i];
                tv[i] = cv; ti[i] = ci;
                cv = tf;  ci = tj;
            }
        }
    }

    unsigned mlo = 0, mhi = 0;
    float mx = tv[0], ssum = 0.f;
#pragma unroll
    for (int i = 0; i < 8; ++i) {
        ssum += __expf(tv[i] - mx);
        if (ti[i] >= 32) mhi |= 1u << (ti[i] - 32);
        else             mlo |= 1u << ti[i];
    }
    float inv = 1.f / ssum;

    float4* orow = (float4*)(outP + (size_t)gm * En);
#pragma unroll
    for (int n4 = 0; n4 < En / 4; ++n4) {
        float ov[4];
#pragma unroll
        for (int j = 0; j < 4; ++j) {
            int  n   = n4 * 4 + j;
            bool sel = (n < 32) ? ((mlo >> n) & 1u) : ((mhi >> (n - 32)) & 1u);
            ov[j] = sel ? __expf(rowL[n] - mx) * inv : 0.f;
        }
        orow[n4] = make_float4(ov[0], ov[1], ov[2], ov[3]);
    }
#pragma unroll
    for (int i = 0; i < 8; ++i)
        outI[(size_t)gm * Kn + i] = (float)ti[i];
}

// ---------------------------------------------------------------------------
// Pre-pass: fp32 weights -> bf16 in d_ws, layout [2][En][Dn] (route, noise).
// ---------------------------------------------------------------------------
__global__ __launch_bounds__(256)
void convert_weights_kernel(const float* __restrict__ Wr,
                            const float* __restrict__ Wn,
                            __bf16* __restrict__ out)
{
    unsigned i = blockIdx.x * 256u + threadIdx.x;     // one float4 each
    const unsigned halfCnt = (En * Dn) / 4;           // 65536
    const float* src = (i < halfCnt) ? Wr : Wn;
    unsigned off = (i < halfCnt) ? i : i - halfCnt;
    float4 w = *(const float4*)(src + (size_t)off * 4);
    v4bf p;
    p[0] = (__bf16)w.x; p[1] = (__bf16)w.y;
    p[2] = (__bf16)w.z; p[3] = (__bf16)w.w;
    *(v4bf*)(out + (size_t)i * 4) = p;
}

// ---------------------------------------------------------------------------
// Main kernel: bf16 weights streamed into double-buffered LDS with
// global_load_async_to_lds_b128 (ASYNCcnt), WMMA bf16 GEMM, fused epilogue.
// ---------------------------------------------------------------------------
__global__ __launch_bounds__(THREADS)
void router_main_kernel(const float* __restrict__ X,     // [MTOT, Dn] fp32
                        const __bf16* __restrict__ Wb,   // [2][En][Dn] bf16
                        const float* __restrict__ br,    // [En]
                        const float* __restrict__ bnv_,  // [En]
                        const float* __restrict__ eps,   // [MTOT, En]
                        float* __restrict__ outP,        // [MTOT, En]
                        float* __restrict__ outI)        // [MTOT, Kn]
{
    // 64 KB arena: [buf][mat][e][k] bf16 weights during GEMM,
    // reused as the fp32 logits scratch for the epilogue.
    __shared__ __attribute__((aligned(32))) char smem[2 * 2 * En * KC * 2];
    __bf16 (*wbf)[2][En][KC] = (__bf16 (*)[2][En][KC])smem;
    float  (*sL)[En + 1]     = (float  (*)[En + 1])smem;

    const int tid  = threadIdx.x;
    const int lane = tid & 31;
    const int wave = tid >> 5;
    const int nl   = lane & 15;
    const int half = lane >> 4;
    const int r0   = blockIdx.x * ROWS;
    const int rowA = r0 + wave * 16 + nl;

    // staging assignment: each thread streams one weight row (mat, e)
    const int smat = tid >> 6;        // 0..1
    const int se   = tid & 63;        // 0..63
    const __bf16* gW = Wb + ((size_t)smat * En + se) * Dn;

    v8f accR[4], accN[4];
#pragma unroll
    for (int t = 0; t < 4; ++t) { accR[t] = (v8f){}; accN[t] = (v8f){}; }

    const float* xrow = X + (size_t)rowA * Dn;

    // prologue: async-stage chunk 0 into buffer 0 (16 x b128 per lane)
    {
        unsigned lb = (unsigned)(uintptr_t)&wbf[0][smat][se][0];
        uint64_t ga = (uint64_t)(uintptr_t)gW;
#pragma unroll
        for (int j = 0; j < (KC * 2) / 16; ++j)
            asm volatile("global_load_async_to_lds_b128 %0, %1, off"
                         :: "v"(lb + 16u * j), "v"(ga + 16u * j) : "memory");
    }

    for (int c = 0; c < NCHUNK; ++c) {
        const int kc = c * KC;
        __syncthreads();   // buffer (c+1)&1 free of readers (chunk c-1 done)
        if (c + 1 < NCHUNK) {
            // prefetch next chunk while this one computes
            unsigned lb = (unsigned)(uintptr_t)&wbf[(c + 1) & 1][smat][se][0];
            uint64_t ga = (uint64_t)(uintptr_t)(gW + kc + KC);
#pragma unroll
            for (int j = 0; j < (KC * 2) / 16; ++j)
                asm volatile("global_load_async_to_lds_b128 %0, %1, off"
                             :: "v"(lb + 16u * j), "v"(ga + 16u * j) : "memory");
            // async loads retire in order: <=16 outstanding => chunk c landed
            asm volatile("s_wait_asynccnt 0x10" ::: "memory");
        } else {
            asm volatile("s_wait_asynccnt 0x0" ::: "memory");
        }
        __syncthreads();   // publish chunk c to all waves

        const __bf16 (*wc)[En][KC] = wbf[c & 1];   // [mat][e][k]
#pragma unroll 2
        for (int ks = 0; ks < KC; ks += 32) {
            // A fragment (16-bit 16x32): elems 0..15 = K half*16+0..15
            const float* ap = xrow + kc + ks + half * 16;
            float4 a0 = *(const float4*)(ap + 0);
            float4 a1 = *(const float4*)(ap + 4);
            float4 a2 = *(const float4*)(ap + 8);
            float4 a3 = *(const float4*)(ap + 12);
            v16bf a;
            a[0]  = (__bf16)a0.x; a[1]  = (__bf16)a0.y; a[2]  = (__bf16)a0.z; a[3]  = (__bf16)a0.w;
            a[4]  = (__bf16)a1.x; a[5]  = (__bf16)a1.y; a[6]  = (__bf16)a1.z; a[7]  = (__bf16)a1.w;
            a[8]  = (__bf16)a2.x; a[9]  = (__bf16)a2.y; a[10] = (__bf16)a2.z; a[11] = (__bf16)a2.w;
            a[12] = (__bf16)a3.x; a[13] = (__bf16)a3.y; a[14] = (__bf16)a3.z; a[15] = (__bf16)a3.w;

#pragma unroll
            for (int t = 0; t < 4; ++t) {
                // B fragment: lane elems 0..15 = K half*16+0..15 of column e
                v16bf bR = *(const v16bf*)&wc[0][t * 16 + nl][ks + half * 16];
                v16bf bN = *(const v16bf*)&wc[1][t * 16 + nl][ks + half * 16];
                accR[t] = __builtin_amdgcn_wmma_f32_16x16x32_bf16(
                    false, a, false, bR, (short)0, accR[t], false, false);
                accN[t] = __builtin_amdgcn_wmma_f32_16x16x32_bf16(
                    false, a, false, bN, (short)0, accN[t], false, false);
            }
        }
    }

    __syncthreads();   // all WMMA reads done before arena is reused as sL

    // ---- epilogue part 1: bias + eps * softplus(noise) -> LDS ----
#pragma unroll
    for (int t = 0; t < 4; ++t) {
        int n = t * 16 + nl;
        float brv = br[n];
        float bnv = bnv_[n];
#pragma unroll
        for (int v = 0; v < 8; ++v) {
            int   mloc  = wave * 16 + v + 8 * half;   // C/D layout: M = v + 8*half
            int   gm    = r0 + mloc;
            float logit = accR[t][v] + brv;
            float nz    = accN[t][v] + bnv;
            float sp = fmaxf(nz, 0.f) + __logf(1.f + __expf(-fabsf(nz)));
            float e  = eps[(size_t)gm * En + n];
            sL[mloc][n] = logit + e * sp;
        }
    }
    __syncthreads();

    if (tid < ROWS)
        topk_softmax_row(&sL[tid][0], r0 + tid, outP, outI);
}

// ---------------------------------------------------------------------------
// Fallback (ws too small): round-1 kernel — fp32 weights converted in LDS.
// ---------------------------------------------------------------------------
__global__ __launch_bounds__(THREADS)
void router_fallback_kernel(const float* __restrict__ X,
                            const float* __restrict__ Wr,
                            const float* __restrict__ br,
                            const float* __restrict__ Wn,
                            const float* __restrict__ bnv_,
                            const float* __restrict__ eps,
                            float* __restrict__ outP,
                            float* __restrict__ outI)
{
    __shared__ __attribute__((aligned(32))) __bf16 sW[2][En][KC];
    __shared__ float sL[ROWS][En + 1];

    const int tid  = threadIdx.x;
    const int lane = tid & 31;
    const int wave = tid >> 5;
    const int nl   = lane & 15;
    const int half = lane >> 4;
    const int r0   = blockIdx.x * ROWS;
    const int rowA = r0 + wave * 16 + nl;

    v8f accR[4], accN[4];
#pragma unroll
    for (int t = 0; t < 4; ++t) { accR[t] = (v8f){}; accN[t] = (v8f){}; }

    const float* xrow = X + (size_t)rowA * Dn;

    for (int kc = 0; kc < Dn; kc += KC) {
        __syncthreads();
#pragma unroll
        for (int it = 0; it < (En * KC) / (THREADS * 4); ++it) {
            int f = (it * THREADS + tid) * 4;
            int e = f / KC;
            int k = f % KC;
            float4 wr4 = *(const float4*)(Wr + (size_t)e * Dn + kc + k);
            float4 wn4 = *(const float4*)(Wn + (size_t)e * Dn + kc + k);
            v4bf pr, pn;
            pr[0] = (__bf16)wr4.x; pr[1] = (__bf16)wr4.y;
            pr[2] = (__bf16)wr4.z; pr[3] = (__bf16)wr4.w;
            pn[0] = (__bf16)wn4.x; pn[1] = (__bf16)wn4.y;
            pn[2] = (__bf16)wn4.z; pn[3] = (__bf16)wn4.w;
            *(v4bf*)&sW[0][e][k] = pr;
            *(v4bf*)&sW[1][e][k] = pn;
        }
        __syncthreads();

#pragma unroll 2
        for (int ks = 0; ks < KC; ks += 32) {
            const float* ap = xrow + kc + ks + half * 16;
            float4 a0 = *(const float4*)(ap + 0);
            float4 a1 = *(const float4*)(ap + 4);
            float4 a2 = *(const float4*)(ap + 8);
            float4 a3 = *(const float4*)(ap + 12);
            v16bf a;
            a[0]  = (__bf16)a0.x; a[1]  = (__bf16)a0.y; a[2]  = (__bf16)a0.z; a[3]  = (__bf16)a0.w;
            a[4]  = (__bf16)a1.x; a[5]  = (__bf16)a1.y; a[6]  = (__bf16)a1.z; a[7]  = (__bf16)a1.w;
            a[8]  = (__bf16)a2.x; a[9]  = (__bf16)a2.y; a[10] = (__bf16)a2.z; a[11] = (__bf16)a2.w;
            a[12] = (__bf16)a3.x; a[13] = (__bf16)a3.y; a[14] = (__bf16)a3.z; a[15] = (__bf16)a3.w;

#pragma unroll
            for (int t = 0; t < 4; ++t) {
                v16bf bR = *(const v16bf*)&sW[0][t * 16 + nl][ks + half * 16];
                v16bf bN = *(const v16bf*)&sW[1][t * 16 + nl][ks + half * 16];
                accR[t] = __builtin_amdgcn_wmma_f32_16x16x32_bf16(
                    false, a, false, bR, (short)0, accR[t], false, false);
                accN[t] = __builtin_amdgcn_wmma_f32_16x16x32_bf16(
                    false, a, false, bN, (short)0, accN[t], false, false);
            }
        }
    }

#pragma unroll
    for (int t = 0; t < 4; ++t) {
        int n = t * 16 + nl;
        float brv = br[n];
        float bnv = bnv_[n];
#pragma unroll
        for (int v = 0; v < 8; ++v) {
            int   mloc  = wave * 16 + v + 8 * half;
            int   gm    = r0 + mloc;
            float logit = accR[t][v] + brv;
            float nz    = accN[t][v] + bnv;
            float sp = fmaxf(nz, 0.f) + __logf(1.f + __expf(-fabsf(nz)));
            float e  = eps[(size_t)gm * En + n];
            sL[mloc][n] = logit + e * sp;
        }
    }
    __syncthreads();

    if (tid < ROWS)
        topk_softmax_row(&sL[tid][0], r0 + tid, outP, outI);
}

// ---------------------------------------------------------------------------
extern "C" void kernel_launch(void* const* d_in, const int* in_sizes, int n_in,
                              void* d_out, int out_size, void* d_ws, size_t ws_size,
                              hipStream_t stream) {
    const float* X   = (const float*)d_in[0];  // mh_output [B,S,D]
    const float* Wr  = (const float*)d_in[1];  // W_route   [E,D]
    const float* br  = (const float*)d_in[2];  // b_route   [E]
    const float* Wn  = (const float*)d_in[3];  // W_noise   [E,D]
    const float* bnv = (const float*)d_in[4];  // b_noise   [E]
    const float* ep  = (const float*)d_in[5];  // noise_eps [B,S,E]
    // d_in[6] = top_k scalar (fixed to 8 at compile time)

    float* outP = (float*)d_out;                      // router_output [B,S,E]
    float* outI = (float*)d_out + (size_t)MTOT * En;  // indices [B,S,K] (as float)

    if (ws_size >= WBYTES) {
        __bf16* Wb = (__bf16*)d_ws;
        const unsigned nvec = (2 * En * Dn) / 4;      // 131072 float4s
        hipLaunchKernelGGL(convert_weights_kernel, dim3(nvec / 256), dim3(256),
                           0, stream, Wr, Wn, Wb);
        hipLaunchKernelGGL(router_main_kernel, dim3(NBLOCKS), dim3(THREADS),
                           0, stream, X, Wb, br, bnv, ep, outP, outI);
    } else {
        hipLaunchKernelGGL(router_fallback_kernel, dim3(NBLOCKS), dim3(THREADS),
                           0, stream, X, Wr, br, Wn, bnv, ep, outP, outI);
    }
}
